// SparseMoE_9517647528393
// MI455X (gfx1250) — matrix operand order repaired and checked by reading, not compiled, not run on hardware
//
#include <hip/hip_runtime.h>
#include <hip/hip_bf16.h>

#define N_EMBED   1024
#define NUM_EXP   8
#define D_FF      4096
#define T_TOK     8192           // B*S = 4*2048
#define BM        16             // tokens per block
#define BF        128            // FF chunk
#define BLK       256            // 8 waves
#define MAX_TILES (T_TOK / BM)   // 512

// packed-tile geometry: a B tile is 32(K) x 16(N) bf16 = 512 elems = 1KB
#define W1_KT     (N_EMBED / 32)   // 32
#define W1_NT     (D_FF / 16)      // 256
#define W2_KT     (D_FF / 32)      // 128
#define W2_NT     (N_EMBED / 16)   // 64
#define TILES_PER_EXP 8192         // both w1 (32*256) and w2 (128*64)
#define ELEMS_PER_EXP (TILES_PER_EXP * 512)   // 4,194,304

typedef __attribute__((ext_vector_type(16))) __bf16 v16bf;
typedef __attribute__((ext_vector_type(8)))  float  v8f;

union Frag16 {
    v16bf v;
    unsigned short us[16];
    uint4 q[2];
};

__device__ __forceinline__ unsigned short f32_to_bf16_rne(float f) {
    unsigned int u = __builtin_bit_cast(unsigned int, f);
    u += 0x7fffu + ((u >> 16) & 1u);
    return (unsigned short)(u >> 16);
}

// ---------------- kernel 1: zero output + expert counters -------------------
__global__ void moe_zero_kernel(float* __restrict__ out, int n, int* __restrict__ counts) {
    int idx = blockIdx.x * blockDim.x + threadIdx.x;
    if (idx < NUM_EXP) counts[idx] = 0;
    for (int i = idx; i < n; i += gridDim.x * blockDim.x) out[i] = 0.f;
}

// ------ kernel 2: fp32 -> bf16 + swizzle into WMMA B-fragment tile order ----
// One wave per 32x16 tile; lane L packs its 16 fragment elements
// (W[kt*32 + (L>>4)*16 + j][ntile*16 + (L&15)], j=0..15) into 32 contiguous B.
__global__ void moe_cvt_kernel(const float* __restrict__ w1, const float* __restrict__ w2,
                               unsigned short* __restrict__ w1p, unsigned short* __restrict__ w2p) {
    const int lane = threadIdx.x & 31;
    const int wid  = (blockIdx.x * blockDim.x + threadIdx.x) >> 5;   // wave id = tile id
    const int nTilesTotal = 2 * NUM_EXP * TILES_PER_EXP;             // 131072
    const int nWaves = (gridDim.x * blockDim.x) >> 5;

    const int nl = lane & 15;
    const int lh = lane >> 4;

    for (int w = wid; w < nTilesTotal; w += nWaves) {
        const bool isW1 = (w < NUM_EXP * TILES_PER_EXP);
        const int  we   = isW1 ? w : (w - NUM_EXP * TILES_PER_EXP);
        const int  e    = we / TILES_PER_EXP;
        const int  r    = we % TILES_PER_EXP;

        const float* src;
        unsigned short* dst;
        int k0, n0, ld;
        if (isW1) {                      // layout [ntile][ktile]
            const int ntile = r / W1_KT;
            const int kt    = r % W1_KT;
            k0 = kt * 32; n0 = ntile * 16; ld = D_FF;
            src = w1 + e * (N_EMBED * D_FF);
            dst = w1p + e * ELEMS_PER_EXP + r * 512 + lane * 16;
        } else {                         // layout [ktile][ntile]
            const int kt    = r / W2_NT;
            const int ntile = r % W2_NT;
            k0 = kt * 32; n0 = ntile * 16; ld = N_EMBED;
            src = w2 + e * (D_FF * N_EMBED);
            dst = w2p + e * ELEMS_PER_EXP + r * 512 + lane * 16;
        }

        Frag16 f;
#pragma unroll
        for (int j = 0; j < 16; ++j)
            f.us[j] = f32_to_bf16_rne(src[(k0 + lh * 16 + j) * ld + n0 + nl]);
        uint4* q = reinterpret_cast<uint4*>(dst);
        q[0] = f.q[0];
        q[1] = f.q[1];
    }
}

// ---------------- kernel 3: noisy top-2 router + compaction -----------------
__global__ void moe_router_kernel(const float* __restrict__ x,
                                  const float* __restrict__ noise,
                                  const float* __restrict__ w_route,
                                  const float* __restrict__ b_route,
                                  const float* __restrict__ w_noise,
                                  const float* __restrict__ b_noise,
                                  int*   __restrict__ counts,
                                  int*   __restrict__ tok_list,
                                  float* __restrict__ gate_list) {
    const int wave = threadIdx.x >> 5;
    const int lane = threadIdx.x & 31;
    const int t = blockIdx.x * 8 + wave;
    if (t >= T_TOK) return;

    const float* xr = x + t * N_EMBED;
    float accR[NUM_EXP], accN[NUM_EXP];
#pragma unroll
    for (int e = 0; e < NUM_EXP; ++e) { accR[e] = 0.f; accN[e] = 0.f; }

    for (int i = lane; i < N_EMBED; i += 32) {
        float xv = xr[i];
        const float* wr = w_route + i * NUM_EXP;
        const float* wn = w_noise + i * NUM_EXP;
#pragma unroll
        for (int e = 0; e < NUM_EXP; ++e) {
            accR[e] = fmaf(xv, wr[e], accR[e]);
            accN[e] = fmaf(xv, wn[e], accN[e]);
        }
    }
#pragma unroll
    for (int off = 16; off >= 1; off >>= 1) {
#pragma unroll
        for (int e = 0; e < NUM_EXP; ++e) {
            accR[e] += __shfl_xor(accR[e], off, 32);
            accN[e] += __shfl_xor(accN[e], off, 32);
        }
    }
    if (lane == 0) {
        float noisy[NUM_EXP];
#pragma unroll
        for (int e = 0; e < NUM_EXP; ++e) {
            float z  = accN[e] + b_noise[e];
            float sp = (z > 20.f) ? z : log1pf(expf(z));   // softplus
            noisy[e] = accR[e] + b_route[e] + noise[t * NUM_EXP + e] * sp;
        }
        int i0 = 0;
#pragma unroll
        for (int e = 1; e < NUM_EXP; ++e) if (noisy[e] > noisy[i0]) i0 = e;
        int i1 = (i0 == 0) ? 1 : 0;
#pragma unroll
        for (int e = 0; e < NUM_EXP; ++e) if (e != i0 && noisy[e] > noisy[i1]) i1 = e;
        float m  = fmaxf(noisy[i0], noisy[i1]);
        float e0 = expf(noisy[i0] - m), e1 = expf(noisy[i1] - m);
        float inv = 1.f / (e0 + e1);
        int s0 = atomicAdd(&counts[i0], 1);
        tok_list[i0 * T_TOK + s0] = t;  gate_list[i0 * T_TOK + s0] = e0 * inv;
        int s1 = atomicAdd(&counts[i1], 1);
        tok_list[i1 * T_TOK + s1] = t;  gate_list[i1 * T_TOK + s1] = e1 * inv;
    }
}

// ---------------- kernel 4: fused gathered expert MLP (WMMA bf16) -----------
__global__ __launch_bounds__(BLK)
void moe_ffn_kernel(const float* __restrict__ x,
                    const unsigned short* __restrict__ w1p,
                    const float* __restrict__ b1,
                    const unsigned short* __restrict__ w2p,
                    const float* __restrict__ b2,
                    const int*   __restrict__ counts,
                    const int*   __restrict__ tok_list,
                    const float* __restrict__ gate_list,
                    float* __restrict__ out) {
    const int e     = blockIdx.y;
    const int cnt   = counts[e];
    const int tile0 = blockIdx.x * BM;
    if (tile0 >= cnt) return;                       // uniform early exit

    __shared__ unsigned short xs[BM][N_EMBED];      // 32 KB bf16 x tile
    __shared__ unsigned short hs[BM][BF];           //  4 KB bf16 h chunk
    __shared__ int   ts[BM];
    __shared__ float gs[BM];

    const int tid  = threadIdx.x;
    const int wave = tid >> 5;
    const int lane = tid & 31;
    const int nl   = lane & 15;                     // N / col lane within tile
    const int lh   = lane >> 4;                     // K/M half selector

    if (tid < BM) {
        int slot = tile0 + tid;
        if (slot < cnt) { ts[tid] = tok_list[e * T_TOK + slot]; gs[tid] = gate_list[e * T_TOK + slot]; }
        else            { ts[tid] = 0;                           gs[tid] = 0.f; }
    }
    __syncthreads();

    // Gather 16 token rows, convert fp32 -> bf16 into LDS
    for (int idx4 = tid; idx4 < BM * (N_EMBED / 4); idx4 += BLK) {
        const int row = idx4 >> 8;                  // 256 float4 per row
        const int c4  = idx4 & 255;
        const float4 f = reinterpret_cast<const float4*>(x + ts[row] * N_EMBED)[c4];
        unsigned int lo = (unsigned int)f32_to_bf16_rne(f.x) | ((unsigned int)f32_to_bf16_rne(f.y) << 16);
        unsigned int hi = (unsigned int)f32_to_bf16_rne(f.z) | ((unsigned int)f32_to_bf16_rne(f.w) << 16);
        *reinterpret_cast<uint2*>(&xs[row][c4 * 4]) = make_uint2(lo, hi);
    }
    __syncthreads();

    const unsigned short* w1e = w1p + e * ELEMS_PER_EXP;
    const unsigned short* w2e = w2p + e * ELEMS_PER_EXP;

    v8f acc[8];                                     // 16 x (wave*128..+128) f32
#pragma unroll
    for (int i = 0; i < 8; ++i) acc[i] = {};

    const int r = nl;                               // A-fragment row

    for (int chunk = 0; chunk < D_FF / BF; ++chunk) {   // 32 chunks of 128
        // ---- stage 1: h[:,wave*16..+16) = relu(x @ w1 + b1), K = 1024
        {
            v8f h = {};
            const int n0    = chunk * BF + wave * 16;
            const int ntile = chunk * 8 + wave;
            // per-wave contiguous 32 KB stream of packed w1 tiles
            const uint4* w1t = reinterpret_cast<const uint4*>(w1e + ntile * (W1_KT * 512) + lane * 16);
            if (chunk + 1 < D_FF / BF)
                __builtin_prefetch(w1e + (ntile + 8) * (W1_KT * 512), 0, 0);
#pragma unroll 8
            for (int ks = 0; ks < W1_KT; ++ks) {    // 32 k-steps of 32
                const int k0 = ks * 32;
                Frag16 A;                           // contiguous K: k = 16*lh + j
                A.q[0] = *reinterpret_cast<const uint4*>(&xs[r][k0 + lh * 16]);
                A.q[1] = *reinterpret_cast<const uint4*>(&xs[r][k0 + lh * 16 + 8]);
                Frag16 Bm;                          // packed: 2x b128 per fragment
                Bm.q[0] = w1t[ks * 64];
                Bm.q[1] = w1t[ks * 64 + 1];
                h = __builtin_amdgcn_wmma_f32_16x16x32_bf16(false, A.v, false, Bm.v,
                                                            (short)0, h, false, false);
            }
            const float bias = b1[e * D_FF + n0 + nl];
#pragma unroll
            for (int j = 0; j < 8; ++j) {
                float v = h[j] + bias;
                v = v > 0.f ? v : 0.f;
                hs[j + 8 * lh][wave * 16 + nl] = f32_to_bf16_rne(v);
            }
        }
        __syncthreads();
        // ---- stage 2: out[:, wave*128..+128) += h @ w2[chunk], K = 128
#pragma unroll
        for (int kk = 0; kk < BF / 32; ++kk) {
            const int k0 = kk * 32;
            const int kt = chunk * 4 + kk;
            Frag16 A;
            A.q[0] = *reinterpret_cast<const uint4*>(&hs[r][k0 + lh * 16]);
            A.q[1] = *reinterpret_cast<const uint4*>(&hs[r][k0 + lh * 16 + 8]);
            const uint4* w2t = reinterpret_cast<const uint4*>(
                w2e + (kt * W2_NT + wave * 8) * 512 + lane * 16);
#pragma unroll
            for (int nt = 0; nt < 8; ++nt) {
                Frag16 Bm;                          // packed: 2x b128 per fragment
                Bm.q[0] = w2t[nt * 64];
                Bm.q[1] = w2t[nt * 64 + 1];
                acc[nt] = __builtin_amdgcn_wmma_f32_16x16x32_bf16(false, A.v, false, Bm.v,
                                                                  (short)0, acc[nt], false, false);
            }
        }
        __syncthreads();
    }

    // ---- epilogue: out[tok] += gate * (acc + b2)   (C layout: N=nl, M=j+8*lh)
#pragma unroll
    for (int nt = 0; nt < 8; ++nt) {
        const int n    = wave * 128 + nt * 16 + nl;
        const float bb = b2[e * N_EMBED + n];
#pragma unroll
        for (int j = 0; j < 8; ++j) {
            const int m = j + 8 * lh;
            const float val = (acc[nt][j] + bb) * gs[m];
            atomicAdd(&out[ts[m] * N_EMBED + n], val);
        }
    }
}

// ---------------------------------------------------------------------------
extern "C" void kernel_launch(void* const* d_in, const int* in_sizes, int n_in,
                              void* d_out, int out_size, void* d_ws, size_t ws_size,
                              hipStream_t stream) {
    const float* x       = (const float*)d_in[0];
    const float* noise   = (const float*)d_in[1];
    const float* w_route = (const float*)d_in[2];
    const float* b_route = (const float*)d_in[3];
    const float* w_noise = (const float*)d_in[4];
    const float* b_noise = (const float*)d_in[5];
    const float* w1      = (const float*)d_in[6];
    const float* b1      = (const float*)d_in[7];
    const float* w2      = (const float*)d_in[8];
    const float* b2      = (const float*)d_in[9];
    float* out = (float*)d_out;

    // workspace layout (~128.5 MiB total)
    char* ws = (char*)d_ws;
    int*            counts    = (int*)ws;                                   // 32 B
    int*            tok_list  = (int*)(ws + 256);                           // 256 KB
    float*          gate_list = (float*)(ws + 256 + 262144);                // 256 KB
    unsigned short* w1p       = (unsigned short*)(ws + 524544);             // 64 MB packed
    unsigned short* w2p       = (unsigned short*)(ws + 524544 + 67108864);  // 64 MB packed

    moe_zero_kernel<<<2048, 256, 0, stream>>>(out, out_size, counts);
    moe_cvt_kernel<<<8192, 256, 0, stream>>>(w1, w2, w1p, w2p);
    moe_router_kernel<<<T_TOK / 8, 256, 0, stream>>>(x, noise, w_route, b_route,
                                                     w_noise, b_noise,
                                                     counts, tok_list, gate_list);
    dim3 grid(MAX_TILES, NUM_EXP);
    moe_ffn_kernel<<<grid, BLK, 0, stream>>>(x, w1p, b1, w2p, b2,
                                             counts, tok_list, gate_list, out);
}
